// EgoPoseFormerHeatmapMVFEX_49452253446269
// MI455X (gfx1250) — matrix operand order repaired and checked by reading, not compiled, not run on hardware
//
#include <hip/hip_runtime.h>
#include <hip/hip_bf16.h>
#include <math.h>
#include <stdint.h>

// ---------------- problem constants ----------------
#define Bb    32
#define Vv    2
#define Jj    15
#define Cc    256
#define HEADS 8
#define Pp    16
#define Dh    32
#define Hf    64
#define Wf    64
#define HW    (Hf * Wf)
#define BVn   (Bb * Vv)
#define FFNd  1024
#define BJ    (Bb * Jj)      // 480
#define BVJ   (BVn * Jj)     // 960

typedef __attribute__((ext_vector_type(16))) __bf16 v16bf;
typedef __attribute__((ext_vector_type(8)))  float  v8f;

union FragBF { uint4 q[2]; v16bf v; };

__device__ __forceinline__ unsigned short f2bf(float f) {
  unsigned int u = __float_as_uint(f);
  unsigned int r = u + 0x7FFFu + ((u >> 16) & 1u);   // round-to-nearest-even
  return (unsigned short)(r >> 16);
}
__device__ __forceinline__ float bf2f(unsigned short u) {
  return __uint_as_float(((unsigned int)u) << 16);
}

// 16-byte async global->LDS copy (CDNA5 GLOBAL_LOAD_ASYNC_TO_LDS_B128,
// tracked by ASYNCcnt).  lds_ptr low 32 bits are the LDS byte offset.
__device__ __forceinline__ void async_copy16(void* lds_ptr, const void* gptr) {
  unsigned lds_off = (unsigned)(uintptr_t)lds_ptr;
  unsigned long long ga = (unsigned long long)(uintptr_t)gptr;
  asm volatile("global_load_async_to_lds_b128 %0, %1, off"
               :: "v"(lds_off), "v"(ga) : "memory");
}
__device__ __forceinline__ void wait_async() {
  asm volatile("s_wait_asynccnt 0" ::: "memory");
}

// =====================================================================
// weight prep: Wt[n][k] (bf16) = W[k][n] (fp32)
// =====================================================================
__global__ __launch_bounds__(256)
void wt_kernel(const float* __restrict__ W, unsigned short* __restrict__ Wt,
               int K, int N)
{
  int idx = blockIdx.x * 256 + threadIdx.x;
  if (idx >= K * N) return;
  int k = idx / N, n = idx % N;
  Wt[(size_t)n * K + k] = f2bf(W[idx]);
}

// =====================================================================
// Wide GEMM:  Out[M,N] = A[M,K] @ W + bias, W given transposed bf16
// Wt[N][K].  Block tile 64 x 256 (A read exactly once when N==256),
// 8 waves (4M x 2N), 8 accumulators / wave -> 8 WMMA per K-step.
// Double-buffered LDS: next K-tile staged (sync A convert + async B
// copies, ASYNCcnt) while current tile feeds the WMMAs.
// Requires N % 256 == 0, K % 32 == 0.
// =====================================================================
__global__ __launch_bounds__(256)
void gemm_wmma_n256(const float* __restrict__ A, const unsigned short* __restrict__ Wt,
                    const float* __restrict__ bias, void* __restrict__ Out,
                    int M, int N, int K, int relu, int out_bf16)
{
  __shared__ unsigned short sA[2][64][32];    // [buf][m][k] bf16
  __shared__ unsigned short sB[2][256][32];   // [buf][n][k] bf16

  const int t    = threadIdx.x;
  const int lane = t & 31;
  const int wave = t >> 5;
  const int wm   = wave & 3;       // m base 16*wm
  const int wn   = wave >> 2;      // n base 128*wn
  const int m0   = blockIdx.y * 64;
  const int n0   = blockIdx.x * 256;
  const int half = lane >> 4;
  const int rA   = lane & 15;

  v8f acc[8];
#pragma unroll
  for (int i = 0; i < 8; ++i)
#pragma unroll
    for (int r = 0; r < 8; ++r) acc[i][r] = 0.f;

  auto stageA = [&](int buf, int k0) {
#pragma unroll
    for (int i = 0; i < 8; ++i) {
      int idx = t + i * 256;
      int r = idx >> 5, c = idx & 31;
      int rg = m0 + r; if (rg >= M) rg = M - 1;
      sA[buf][r][c] = f2bf(A[(size_t)rg * K + (k0 + c)]);
    }
  };
  auto stageB = [&](int buf, int k0) {
#pragma unroll
    for (int i = 0; i < 4; ++i) {
      int c = t + i * 256;            // 1024 chunks of 16B
      int n = c >> 2;
      int koff = (c & 3) * 8;
      async_copy16(&sB[buf][n][koff], Wt + (size_t)(n0 + n) * K + k0 + koff);
    }
  };

  stageA(0, 0);
  stageB(0, 0);

  for (int k0 = 0; k0 < K; k0 += 32) {
    const int cur = (k0 >> 5) & 1;
    wait_async();          // current buffer's async B copies done
    __syncthreads();       // + A converts visible; alt buffer free (WAR safe)

    if (k0 + 32 < K) {     // prefetch next K-tile into alternate buffer
      stageA(cur ^ 1, k0 + 32);
      stageB(cur ^ 1, k0 + 32);
    }

    // A fragment: lane(half=0) K{0..15}; (half=1) K{16..31}
    FragBF fa;
    const uint4* pa = (const uint4*)&sA[cur][wm * 16 + rA][0];
    fa.q[0] = pa[half * 2];
    fa.q[1] = pa[half * 2 + 1];

    // all 8 B fragments into distinct registers -> staggered DS waits
    FragBF fb[8];
#pragma unroll
    for (int nt = 0; nt < 8; ++nt) {
      const uint4* pb = (const uint4*)&sB[cur][wn * 128 + nt * 16 + rA][0];
      fb[nt].q[0] = pb[half * 2];
      fb[nt].q[1] = pb[half * 2 + 1];
    }
#pragma unroll
    for (int nt = 0; nt < 8; ++nt)
      acc[nt] = __builtin_amdgcn_wmma_f32_16x16x32_bf16(false, fa.v, false, fb[nt].v,
                                                        (short)0, acc[nt], false, false);
  }

  // ---- store (D layout: lane -> N, vgpr r + 8*half -> M) ----
#pragma unroll
  for (int nt = 0; nt < 8; ++nt) {
#pragma unroll
    for (int r = 0; r < 8; ++r) {
      int mrow = m0 + wm * 16 + half * 8 + r;
      if (mrow >= M) continue;
      int col = n0 + wn * 128 + nt * 16 + rA;
      float v = acc[nt][r] + bias[col];
      if (relu) v = fmaxf(v, 0.f);
      if (out_bf16) ((unsigned short*)Out)[(size_t)mrow * N + col] = f2bf(v);
      else          ((float*)Out)[(size_t)mrow * N + col] = v;
    }
  }
}

// =====================================================================
// Narrow GEMM (64x64 tile) for N % 64 == 0 cases (attn logits, N=128).
// Same bf16/WMMA scheme, Wt pre-transposed, async B staging.
// =====================================================================
__global__ __launch_bounds__(256)
void gemm_wmma_n64(const float* __restrict__ A, const unsigned short* __restrict__ Wt,
                   const float* __restrict__ bias, void* __restrict__ Out,
                   int M, int N, int K, int relu, int out_bf16)
{
  __shared__ unsigned short sA[64][32];
  __shared__ unsigned short sB[64][32];

  const int t    = threadIdx.x;
  const int lane = t & 31;
  const int wave = t >> 5;
  const int wm   = wave & 3;
  const int wn   = wave >> 2;
  const int m0   = blockIdx.y * 64;
  const int n0   = blockIdx.x * 64;
  const int half = lane >> 4;
  const int rA   = lane & 15;

  v8f acc0 = {};
  v8f acc1 = {};

  for (int k0 = 0; k0 < K; k0 += 32) {
#pragma unroll
    for (int i = 0; i < 8; ++i) {
      int idx = t + i * 256;
      int r = idx >> 5, c = idx & 31;
      int rg = m0 + r; if (rg >= M) rg = M - 1;
      sA[r][c] = f2bf(A[(size_t)rg * K + (k0 + c)]);
    }
    {
      int n = t >> 2;                 // 256 chunks, 1 per thread
      int koff = (t & 3) * 8;
      async_copy16(&sB[n][koff], Wt + (size_t)(n0 + n) * K + k0 + koff);
    }
    wait_async();
    __syncthreads();

    FragBF fa, fb0, fb1;
    const uint4* pa = (const uint4*)&sA[wm * 16 + rA][0];
    fa.q[0] = pa[half * 2];
    fa.q[1] = pa[half * 2 + 1];
    const uint4* pb0 = (const uint4*)&sB[wn * 32 + rA][0];
    fb0.q[0] = pb0[half * 2];
    fb0.q[1] = pb0[half * 2 + 1];
    const uint4* pb1 = (const uint4*)&sB[wn * 32 + 16 + rA][0];
    fb1.q[0] = pb1[half * 2];
    fb1.q[1] = pb1[half * 2 + 1];

    acc0 = __builtin_amdgcn_wmma_f32_16x16x32_bf16(false, fa.v, false, fb0.v,
                                                   (short)0, acc0, false, false);
    acc1 = __builtin_amdgcn_wmma_f32_16x16x32_bf16(false, fa.v, false, fb1.v,
                                                   (short)0, acc1, false, false);
    __syncthreads();
  }

#pragma unroll
  for (int r = 0; r < 8; ++r) {
    int mrow = m0 + wm * 16 + half * 8 + r;
    if (mrow >= M) continue;
    int c0 = n0 + wn * 32 + rA;
    float v0 = acc0[r] + bias[c0];
    float v1 = acc1[r] + bias[c0 + 16];
    if (relu) { v0 = fmaxf(v0, 0.f); v1 = fmaxf(v1, 0.f); }
    if (out_bf16) {
      unsigned short* o = (unsigned short*)Out;
      o[(size_t)mrow * N + c0]      = f2bf(v0);
      o[(size_t)mrow * N + c0 + 16] = f2bf(v1);
    } else {
      float* o = (float*)Out;
      o[(size_t)mrow * N + c0]      = v0;
      o[(size_t)mrow * N + c0 + 16] = v1;
    }
  }
}

// =====================================================================
// Deformable sampling + per-(h) softmax over P + attention-weighted sum.
// block = (bv, j), thread = (h, d).  val is bf16 [BV*HW, 256].
// =====================================================================
__global__ __launch_bounds__(256)
void deform_sample_kernel(const unsigned short* __restrict__ val,
                          const float* __restrict__ offb,   // [480,256]
                          const float* __restrict__ attnl,  // [480,128]
                          const float* __restrict__ ref,    // [BV*J,2]
                          float* __restrict__ da_pre)       // [960,256]
{
  const int blk = blockIdx.x;          // bv*J + j
  const int bv = blk / Jj, j = blk % Jj;
  const int b  = bv / Vv;
  const int t  = threadIdx.x;
  const int h  = t >> 5, d = t & 31;
  const int r480 = b * Jj + j;

  const float rx = ref[(size_t)blk * 2 + 0];
  const float ry = ref[(size_t)blk * 2 + 1];

  float mx = -1e30f;
#pragma unroll
  for (int p = 0; p < Pp; ++p)
    mx = fmaxf(mx, attnl[(size_t)r480 * 128 + h * Pp + p]);

  const size_t vbase = (size_t)bv * HW * Cc + h * Dh + d;
  float acc = 0.f, wsum = 0.f;

#pragma unroll
  for (int p = 0; p < Pp; ++p) {
    float w = __expf(attnl[(size_t)r480 * 128 + h * Pp + p] - mx);
    wsum += w;
    float ox = offb[(size_t)r480 * 256 + h * 32 + p * 2 + 0];
    float oy = offb[(size_t)r480 * 256 + h * 32 + p * 2 + 1];
    float x = rx * (float)Wf + ox - 0.5f;
    float y = ry * (float)Hf + oy - 0.5f;
    float x0f = floorf(x), y0f = floorf(y);
    int x0 = (int)x0f, y0 = (int)y0f;
    float wx1 = x - x0f, wx0 = 1.f - wx1;
    float wy1 = y - y0f, wy0 = 1.f - wy1;

    float s = 0.f;
#pragma unroll
    for (int cy = 0; cy < 2; ++cy) {
#pragma unroll
      for (int cx = 0; cx < 2; ++cx) {
        int ix = x0 + cx, iy = y0 + cy;
        float inb = (ix >= 0 && ix < Wf && iy >= 0 && iy < Hf) ? 1.f : 0.f;
        int cix = ix < 0 ? 0 : (ix > Wf - 1 ? Wf - 1 : ix);
        int ciy = iy < 0 ? 0 : (iy > Hf - 1 ? Hf - 1 : iy);
        float sv = bf2f(val[vbase + (size_t)(ciy * Wf + cix) * Cc]) * inb;
        float ww = (cx ? wx1 : wx0) * (cy ? wy1 : wy0);
        s += sv * ww;
      }
    }
    acc += w * s;
  }
  da_pre[(size_t)blk * Cc + h * Dh + d] = acc / wsum;
}

// =====================================================================
// mask by anchors_valid and concat views:  A2[b*J+j][v*C+c]
// =====================================================================
__global__ __launch_bounds__(256)
void mask_concat_kernel(const float* __restrict__ da,
                        const unsigned char* __restrict__ valid,
                        float* __restrict__ A2)
{
  int idx = blockIdx.x * 256 + threadIdx.x;
  if (idx >= BJ * (Vv * Cc)) return;
  int r  = idx >> 9;          // /512
  int ch = idx & 511;
  int v  = ch >> 8;
  int c  = ch & 255;
  int b = r / Jj, j = r % Jj;
  int bvj = (b * Vv + v) * Jj + j;
  float m = valid[bvj] ? 1.f : 0.f;
  A2[idx] = da[(size_t)bvj * Cc + c] * m;
}

// =====================================================================
// out = LayerNorm(a + bsrc) over C=256 ; one row per block
// =====================================================================
__global__ __launch_bounds__(256)
void add_ln_kernel(const float* __restrict__ a, const float* __restrict__ bsrc,
                   const float* __restrict__ g, const float* __restrict__ be,
                   float* __restrict__ out)
{
  __shared__ float r1[256];
  __shared__ float r2[256];
  const int r = blockIdx.x, t = threadIdx.x;
  float x = a[(size_t)r * Cc + t] + bsrc[(size_t)r * Cc + t];
  r1[t] = x; r2[t] = x * x;
  __syncthreads();
  for (int s = 128; s > 0; s >>= 1) {
    if (t < s) { r1[t] += r1[t + s]; r2[t] += r2[t + s]; }
    __syncthreads();
  }
  float mean = r1[0] * (1.f / 256.f);
  float var  = r2[0] * (1.f / 256.f) - mean * mean;
  float inv  = rsqrtf(var + 1e-5f);
  out[(size_t)r * Cc + t] = (x - mean) * inv * g[t] + be[t];
}

// =====================================================================
// tiny spatial MHA over J=15 joints; one block per (b, h)
// =====================================================================
__global__ __launch_bounds__(256)
void mha_kernel(const float* __restrict__ qb, const float* __restrict__ kb,
                const float* __restrict__ vb, float* __restrict__ sa_pre)
{
  __shared__ float qs[Jj][Dh], ks[Jj][Dh], vs[Jj][Dh];
  __shared__ float sc[Jj][Jj];
  const int blk = blockIdx.x;
  const int b = blk / HEADS, h = blk % HEADS;
  const int t = threadIdx.x;

  for (int idx = t; idx < Jj * Dh; idx += 256) {
    int jj = idx / Dh, dd = idx % Dh;
    size_t base = (size_t)(b * Jj + jj) * Cc + h * Dh + dd;
    qs[jj][dd] = qb[base]; ks[jj][dd] = kb[base]; vs[jj][dd] = vb[base];
  }
  __syncthreads();

  const float scale = 0.1767766952966369f;  // 32^-0.5
  for (int idx = t; idx < Jj * Jj; idx += 256) {
    int jq = idx / Jj, jk = idx % Jj;
    float s = 0.f;
#pragma unroll
    for (int dd = 0; dd < Dh; ++dd) s += qs[jq][dd] * ks[jk][dd];
    sc[jq][jk] = s * scale;
  }
  __syncthreads();

  if (t < Jj) {
    float mxv = -1e30f;
    for (int k2 = 0; k2 < Jj; ++k2) mxv = fmaxf(mxv, sc[t][k2]);
    float sm = 0.f;
    for (int k2 = 0; k2 < Jj; ++k2) { float e = __expf(sc[t][k2] - mxv); sc[t][k2] = e; sm += e; }
    float inv = 1.f / sm;
    for (int k2 = 0; k2 < Jj; ++k2) sc[t][k2] *= inv;
  }
  __syncthreads();

  for (int idx = t; idx < Jj * Dh; idx += 256) {
    int jj = idx / Dh, dd = idx % Dh;
    float o = 0.f;
#pragma unroll
    for (int k2 = 0; k2 < Jj; ++k2) o += sc[jj][k2] * vs[k2][dd];
    sa_pre[(size_t)(b * Jj + jj) * Cc + h * Dh + dd] = o;
  }
}

// =====================================================================
// host-side driver
// =====================================================================
static inline void prep_wt(const float* W, unsigned short* Wt, int K, int N,
                           hipStream_t s)
{
  int total = K * N;
  wt_kernel<<<(total + 255) / 256, 256, 0, s>>>(W, Wt, K, N);
}

static inline void gemm256(const float* A, const unsigned short* Wt, const float* bias,
                           void* out, int M, int N, int K, int relu, int obf,
                           hipStream_t s)
{
  dim3 grid(N / 256, (M + 63) / 64);
  gemm_wmma_n256<<<grid, dim3(256), 0, s>>>(A, Wt, bias, out, M, N, K, relu, obf);
}

extern "C" void kernel_launch(void* const* d_in, const int* in_sizes, int n_in,
                              void* d_out, int out_size, void* d_ws, size_t ws_size,
                              hipStream_t stream)
{
  const float* feat_query = (const float*)d_in[0];        // [B*J, C]
  const float* frame_feat = (const float*)d_in[1];        // [BV*HW, C]
  const float* anchors2d  = (const float*)d_in[2];        // [BV*J, 2]
  const unsigned char* avalid = (const unsigned char*)d_in[3];  // bool [BV*J]
  const float* W_value = (const float*)d_in[4];  const float* b_value = (const float*)d_in[5];
  const float* W_off   = (const float*)d_in[6];  const float* b_off   = (const float*)d_in[7];
  const float* W_attn  = (const float*)d_in[8];  const float* b_attn  = (const float*)d_in[9];
  const float* W_out   = (const float*)d_in[10]; const float* b_out   = (const float*)d_in[11];
  const float* W_fuse  = (const float*)d_in[12]; const float* b_fuse  = (const float*)d_in[13];
  const float* W_q = (const float*)d_in[14]; const float* b_q = (const float*)d_in[15];
  const float* W_k = (const float*)d_in[16]; const float* b_k = (const float*)d_in[17];
  const float* W_v = (const float*)d_in[18]; const float* b_v = (const float*)d_in[19];
  const float* W_o = (const float*)d_in[20]; const float* b_o = (const float*)d_in[21];
  const float* W_f1 = (const float*)d_in[22]; const float* b_f1 = (const float*)d_in[23];
  const float* W_f2 = (const float*)d_in[24]; const float* b_f2 = (const float*)d_in[25];
  const float* g_c = (const float*)d_in[26]; const float* be_c = (const float*)d_in[27];
  const float* g_s = (const float*)d_in[28]; const float* be_s = (const float*)d_in[29];
  const float* g_f = (const float*)d_in[30]; const float* be_f = (const float*)d_in[31];

  char* ws = (char*)d_ws;
  size_t cur = 0;
  auto alloc = [&](size_t bytes) -> void* {
    void* p = ws + cur;
    cur += (bytes + 255) & ~(size_t)255;
    return p;
  };

  // activations
  unsigned short* val = (unsigned short*)alloc((size_t)BVn * HW * Cc * 2);  // bf16
  float* offb   = (float*)alloc((size_t)BJ * 256 * 4);
  float* attnb  = (float*)alloc((size_t)BJ * 128 * 4);
  float* da_pre = (float*)alloc((size_t)BVJ * Cc * 4);
  float* da     = (float*)alloc((size_t)BVJ * Cc * 4);
  float* A2     = (float*)alloc((size_t)BJ * Vv * Cc * 4);
  float* fused  = (float*)alloc((size_t)BJ * Cc * 4);
  float* x1     = (float*)alloc((size_t)BJ * Cc * 4);
  float* qbuf   = (float*)alloc((size_t)BJ * Cc * 4);
  float* kbuf   = (float*)alloc((size_t)BJ * Cc * 4);
  float* vbuf   = (float*)alloc((size_t)BJ * Cc * 4);
  float* sa_pre = (float*)alloc((size_t)BJ * Cc * 4);
  float* sa     = (float*)alloc((size_t)BJ * Cc * 4);
  float* x2     = (float*)alloc((size_t)BJ * Cc * 4);
  float* h1     = (float*)alloc((size_t)BJ * FFNd * 4);
  float* ff     = (float*)alloc((size_t)BJ * Cc * 4);

  // pre-transposed bf16 weights  Wt[N][K]
  unsigned short* Wt_value = (unsigned short*)alloc((size_t)256 * 256 * 2);
  unsigned short* Wt_off   = (unsigned short*)alloc((size_t)256 * 256 * 2);
  unsigned short* Wt_attn  = (unsigned short*)alloc((size_t)128 * 256 * 2);
  unsigned short* Wt_out   = (unsigned short*)alloc((size_t)256 * 256 * 2);
  unsigned short* Wt_fuse  = (unsigned short*)alloc((size_t)256 * 512 * 2);
  unsigned short* Wt_q     = (unsigned short*)alloc((size_t)256 * 256 * 2);
  unsigned short* Wt_k     = (unsigned short*)alloc((size_t)256 * 256 * 2);
  unsigned short* Wt_v     = (unsigned short*)alloc((size_t)256 * 256 * 2);
  unsigned short* Wt_o     = (unsigned short*)alloc((size_t)256 * 256 * 2);
  unsigned short* Wt_f1    = (unsigned short*)alloc((size_t)1024 * 256 * 2);
  unsigned short* Wt_f2    = (unsigned short*)alloc((size_t)256 * 1024 * 2);

  // --- weight prep (tiny) ---
  prep_wt(W_value, Wt_value, 256, 256,  stream);
  prep_wt(W_off,   Wt_off,   256, 256,  stream);
  prep_wt(W_attn,  Wt_attn,  256, 128,  stream);
  prep_wt(W_out,   Wt_out,   256, 256,  stream);
  prep_wt(W_fuse,  Wt_fuse,  512, 256,  stream);
  prep_wt(W_q,     Wt_q,     256, 256,  stream);
  prep_wt(W_k,     Wt_k,     256, 256,  stream);
  prep_wt(W_v,     Wt_v,     256, 256,  stream);
  prep_wt(W_o,     Wt_o,     256, 256,  stream);
  prep_wt(W_f1,    Wt_f1,    256, 1024, stream);
  prep_wt(W_f2,    Wt_f2,    1024, 256, stream);

  // 1. value projection (dominant GEMM, bf16 out, A read exactly once)
  gemm256(frame_feat, Wt_value, b_value, val, BVn * HW, Cc, Cc, 0, 1, stream);
  // 2. sampling offsets + attention logits (q depends only on (b,j))
  gemm256(feat_query, Wt_off, b_off, offb, BJ, 256, Cc, 0, 0, stream);
  gemm_wmma_n64<<<dim3(2, (BJ + 63) / 64), dim3(256), 0, stream>>>(
      feat_query, Wt_attn, b_attn, attnb, BJ, 128, Cc, 0, 0);
  // 3. deformable bilinear sampling + softmax(P) + weighted sum
  deform_sample_kernel<<<BVJ, 256, 0, stream>>>(val, offb, attnb, anchors2d, da_pre);
  // 4. output projection of deformable attention
  gemm256(da_pre, Wt_out, b_out, da, BVJ, Cc, Cc, 0, 0, stream);
  // 5. mask invalid anchors + concat views -> [480, 512]
  mask_concat_kernel<<<(BJ * Vv * Cc + 255) / 256, 256, 0, stream>>>(da, avalid, A2);
  // 6. view fusion
  gemm256(A2, Wt_fuse, b_fuse, fused, BJ, Cc, Vv * Cc, 0, 0, stream);
  // 7. x1 = LN(q + fused)
  add_ln_kernel<<<BJ, 256, 0, stream>>>(feat_query, fused, g_c, be_c, x1);
  // 8. spatial MHA projections
  gemm256(x1, Wt_q, b_q, qbuf, BJ, Cc, Cc, 0, 0, stream);
  gemm256(x1, Wt_k, b_k, kbuf, BJ, Cc, Cc, 0, 0, stream);
  gemm256(x1, Wt_v, b_v, vbuf, BJ, Cc, Cc, 0, 0, stream);
  // 9. attention over 15 joints
  mha_kernel<<<Bb * HEADS, 256, 0, stream>>>(qbuf, kbuf, vbuf, sa_pre);
  // 10. output projection + x2 = LN(x1 + sa)
  gemm256(sa_pre, Wt_o, b_o, sa, BJ, Cc, Cc, 0, 0, stream);
  add_ln_kernel<<<BJ, 256, 0, stream>>>(x1, sa, g_s, be_s, x2);
  // 11. FFN (relu fused in GEMM1)
  gemm256(x2, Wt_f1, b_f1, h1, BJ, FFNd, Cc, 1, 0, stream);
  gemm256(h1, Wt_f2, b_f2, ff, BJ, Cc, FFNd, 0, 0, stream);
  // 12. final LN -> d_out
  add_ln_kernel<<<BJ, 256, 0, stream>>>(x2, ff, g_f, be_f, (float*)d_out);
}
